// KVCachedTransformer_60430189855183
// MI455X (gfx1250) — matrix operand order repaired and checked by reading, not compiled, not run on hardware
//
#include <hip/hip_runtime.h>
#include <math.h>

typedef __bf16 bf16;
typedef __attribute__((ext_vector_type(16))) __bf16 v16bf;
typedef __attribute__((ext_vector_type(8)))  __bf16 v8bf;
typedef __attribute__((ext_vector_type(8)))  float  v8f;

#define BATCH 2
#define SEQ   2048
#define DMODEL 1024
#define NHEAD 16
#define DHEAD 64
#define DMLP  4096
#define VOCAB 32000
#define NLAYERS 2
#define BS (BATCH*SEQ)   // 4096

// ---------------- WMMA fragment loaders ----------------
// A-matrix 16x32 bf16 layout (per lane: row m = lane%16, half = lane/16):
//   elems 0..7  = K (half*8 + j),  elems 8..15 = K (16 + half*8 + j)  -> two contiguous 16B chunks
// B-matrix 32x16 bf16 layout (per lane: col n = lane%16, half = lane/16):
//   elems 0..7  = K (half*8 + j),  elems 8..15 = K (16 + half*8 + j)  -> two contiguous 16B chunks
__device__ __forceinline__ v16bf pack2(v8bf lo, v8bf hi) {
  v16bf r;
#pragma unroll
  for (int i = 0; i < 8; ++i) { r[i] = lo[i]; r[i + 8] = hi[i]; }
  return r;
}
__device__ __forceinline__ v16bf load_a_frag(const bf16* A, int lda, int m, int k0, int half) {
  const bf16* p = A + (long)m * lda + k0 + half * 8;
  return pack2(*(const v8bf*)p, *(const v8bf*)(p + 16));
}
__device__ __forceinline__ v16bf load_b_frag(const bf16* BT, int ldbt, int n, int k0, int half) {
  const bf16* p = BT + (long)n * ldbt + k0 + half * 8;
  return pack2(*(const v8bf*)p, *(const v8bf*)(p + 16));
}

// ------------- 16-lane butterfly reductions (DPP8 + permlane16, no LDS) -------------
#define DPP8SEL(a,b,c,d,e,f,g,h) \
  ((unsigned)(a)|((unsigned)(b)<<3)|((unsigned)(c)<<6)|((unsigned)(d)<<9)| \
   ((unsigned)(e)<<12)|((unsigned)(f)<<15)|((unsigned)(g)<<18)|((unsigned)(h)<<21))

template <unsigned SEL>
__device__ __forceinline__ float dppxor(float x) {
  int i = __builtin_bit_cast(int, x);
  i = __builtin_amdgcn_mov_dpp8(i, SEL);
  return __builtin_bit_cast(float, i);
}
// lane i (within 16-lane row) <- lane i^8 of same row
__device__ __forceinline__ float permx8(float x) {
  int i = __builtin_bit_cast(int, x);
  i = __builtin_amdgcn_permlane16(i, i, 0xFEDCBA98u, 0x76543210u, false, false);
  return __builtin_bit_cast(float, i);
}
__device__ __forceinline__ float rmax16(float v) {
  v = fmaxf(v, dppxor<DPP8SEL(1,0,3,2,5,4,7,6)>(v));
  v = fmaxf(v, dppxor<DPP8SEL(2,3,0,1,6,7,4,5)>(v));
  v = fmaxf(v, dppxor<DPP8SEL(4,5,6,7,0,1,2,3)>(v));
  v = fmaxf(v, permx8(v));
  return v;
}
__device__ __forceinline__ float rsum16(float v) {
  v += dppxor<DPP8SEL(1,0,3,2,5,4,7,6)>(v);
  v += dppxor<DPP8SEL(2,3,0,1,6,7,4,5)>(v);
  v += dppxor<DPP8SEL(4,5,6,7,0,1,2,3)>(v);
  v += permx8(v);
  return v;
}

// ---------------- generic WMMA GEMM ----------------
// One wave computes a (MT*16)(M) x 64(N) strip: MT*4 WMMAs per K-step
// (B reuse x MT, A reuse x 4).  A:[M,K] bf16 row-major, BT:[N,K] bf16 row-major.
// grid = (N/64, M/(MT*16), batch).
// Epilogue can emit f32 row-major (outF), bf16 row-major (outB), bf16 transposed (outT).
template <int MT, bool RESID, bool GELU>
__global__ __launch_bounds__(32) void gemm_wmma(
    const bf16* __restrict__ A, int lda, long aBatch,
    const bf16* __restrict__ BT, int ldbt, long bBatch,
    float* outF, bf16* outB, int ldc, long cBatch,
    bf16* outT, int ldt, long cBatchT,
    const float* resid, int K) {
  const int lane = threadIdx.x, half = lane >> 4, ln = lane & 15;
  const int m0 = blockIdx.y * (MT * 16);
  const long n0 = (long)blockIdx.x * 64;
  A  += (long)blockIdx.z * aBatch;
  BT += (long)blockIdx.z * bBatch;
  const long coff  = (long)blockIdx.z * cBatch;
  const long coffT = (long)blockIdx.z * cBatchT;

  v8f acc[MT][4];
#pragma unroll
  for (int mi = 0; mi < MT; ++mi)
#pragma unroll
    for (int t = 0; t < 4; ++t)
#pragma unroll
      for (int r = 0; r < 8; ++r) acc[mi][t][r] = 0.0f;

  for (int k0 = 0; k0 < K; k0 += 32) {
    v16bf af[MT];
#pragma unroll
    for (int mi = 0; mi < MT; ++mi)
      af[mi] = load_a_frag(A, lda, m0 + mi * 16 + ln, k0, half);
#pragma unroll
    for (int t = 0; t < 4; ++t) {
      v16bf bfm = load_b_frag(BT, ldbt, (int)(n0 + t * 16 + ln), k0, half);
#pragma unroll
      for (int mi = 0; mi < MT; ++mi)
        acc[mi][t] = __builtin_amdgcn_wmma_f32_16x16x32_bf16(false, af[mi], false, bfm,
                                                             (short)0, acc[mi][t], false, false);
    }
  }
#pragma unroll
  for (int mi = 0; mi < MT; ++mi)
#pragma unroll
    for (int t = 0; t < 4; ++t)
#pragma unroll
      for (int r = 0; r < 8; ++r) {
        const long row = m0 + mi * 16 + half * 8 + r;
        const long col = n0 + t * 16 + ln;
        float v = acc[mi][t][r];
        if (RESID) v += resid[coff + row * ldc + col];
        if (GELU)  v = 0.5f * v * (1.0f + erff(v * 0.70710678118654752f));
        if (outF) outF[coff + row * ldc + col] = v;
        if (outB) outB[coff + row * ldc + col] = (bf16)v;
        if (outT) outT[coffT + col * ldt + row] = (bf16)v;
      }
}

// ---------------- fused causal flash attention ----------------
// q,k: [B,S,H*Dh] bf16.  xvT: [H][Dh][B*S] bf16 (= (h@V_h)^T).  z out: [B,S,H*Dh] bf16.
// One wave per (q-tile of 16 rows, head, batch).  k-tiles of 32 columns.
__global__ __launch_bounds__(32) void flash_attn_kernel(
    const bf16* __restrict__ q, const bf16* __restrict__ k,
    const bf16* __restrict__ xvT, bf16* __restrict__ z) {
  __shared__ __align__(32) bf16 ptile[16 * 32];
  const int lane = threadIdx.x, half = lane >> 4, ln = lane & 15;
  const int q0 = blockIdx.x * 16, h = blockIdx.y, b = blockIdx.z;
  const int Dm = NHEAD * DHEAD;
  const float SCALE = 0.125f;  // 1/sqrt(64)

  const bf16* qb = q + (long)b * SEQ * Dm + h * DHEAD;
  const bf16* kb = k + (long)b * SEQ * Dm + h * DHEAD;
  const bf16* vb = xvT + (long)h * DHEAD * BS + (long)b * SEQ;

  v16bf qf[2];
  qf[0] = load_a_frag(qb, Dm, q0 + ln, 0, half);
  qf[1] = load_a_frag(qb, Dm, q0 + ln, 32, half);

  v8f o[4];
  float mr[8], lr[8];
#pragma unroll
  for (int t = 0; t < 4; ++t)
#pragma unroll
    for (int r = 0; r < 8; ++r) o[t][r] = 0.0f;
#pragma unroll
  for (int r = 0; r < 8; ++r) { mr[r] = -1.0e30f; lr[r] = 0.0f; }

  const int ntiles = (q0 + 47) >> 5;  // cover columns 0 .. q0+15
  for (int t = 0; t < ntiles; ++t) {
    const int kt0 = t << 5;
    v8f s0, s1;
#pragma unroll
    for (int r = 0; r < 8; ++r) { s0[r] = 0.0f; s1[r] = 0.0f; }
#pragma unroll
    for (int c = 0; c < 2; ++c) {  // Dh chunks of 32
      v16bf kf0 = load_b_frag(kb, Dm, kt0 + ln, c * 32, half);
      s0 = __builtin_amdgcn_wmma_f32_16x16x32_bf16(false, qf[c], false, kf0, (short)0, s0, false, false);
      v16bf kf1 = load_b_frag(kb, Dm, kt0 + 16 + ln, c * 32, half);
      s1 = __builtin_amdgcn_wmma_f32_16x16x32_bf16(false, qf[c], false, kf1, (short)0, s1, false, false);
    }
    // online softmax update, per accumulator row r (row = q0 + half*8 + r)
#pragma unroll
    for (int r = 0; r < 8; ++r) {
      const int row = q0 + half * 8 + r;
      float v0 = (kt0 + ln      <= row) ? s0[r] * SCALE : -3.0e38f;
      float v1 = (kt0 + 16 + ln <= row) ? s1[r] * SCALE : -3.0e38f;
      float tm = rmax16(fmaxf(v0, v1));
      float mn = fmaxf(mr[r], tm);
      float al = __expf(mr[r] - mn);
      float p0 = __expf(v0 - mn), p1 = __expf(v1 - mn);
      lr[r] = lr[r] * al + rsum16(p0 + p1);
      mr[r] = mn;
#pragma unroll
      for (int t4 = 0; t4 < 4; ++t4) o[t4][r] *= al;
      ptile[(half * 8 + r) * 32 + ln]      = (bf16)p0;
      ptile[(half * 8 + r) * 32 + 16 + ln] = (bf16)p1;
    }
    __syncthreads();
    v16bf pf = load_a_frag(ptile, 32, ln, 0, half);
#pragma unroll
    for (int t4 = 0; t4 < 4; ++t4) {
      v16bf vf = load_b_frag(vb, BS, t4 * 16 + ln, kt0, half);
      o[t4] = __builtin_amdgcn_wmma_f32_16x16x32_bf16(false, pf, false, vf, (short)0, o[t4], false, false);
    }
    __syncthreads();
  }
#pragma unroll
  for (int r = 0; r < 8; ++r) {
    const float inv = 1.0f / lr[r];
    const long row = (long)b * SEQ + q0 + half * 8 + r;
#pragma unroll
    for (int t4 = 0; t4 < 4; ++t4)
      z[row * Dm + h * DHEAD + t4 * 16 + ln] = (bf16)(o[t4][r] * inv);
  }
}

// ---------------- helpers ----------------
__global__ void embed_kernel(const int* __restrict__ x, const float* __restrict__ emb,
                             const float* __restrict__ pos, float* __restrict__ h,
                             bf16* __restrict__ hb) {
  const int tokIdx = blockIdx.x;           // b*S + s
  const int s = tokIdx % SEQ;
  const int tok = x[tokIdx];
  const float* e = emb + (long)tok * DMODEL;
  const float* p = pos + (long)s * DMODEL;
  for (int i = threadIdx.x; i < DMODEL; i += blockDim.x) {
    float v = e[i] + p[i];
    h[(long)tokIdx * DMODEL + i] = v;
    hb[(long)tokIdx * DMODEL + i] = (bf16)v;
  }
}

__global__ void cvt_kernel(const float* __restrict__ in, bf16* __restrict__ out, long n) {
  long i = (long)blockIdx.x * blockDim.x + threadIdx.x;
  const long stride = (long)gridDim.x * blockDim.x;
  for (; i < n; i += stride) out[i] = (bf16)in[i];
}

// in: [batch][R][C] f32 -> out: [batch][C][R] bf16.  grid.x*256 == R*C, grid.z == batch
__global__ void tcvt_kernel(const float* __restrict__ in, bf16* __restrict__ out,
                            int R, int C, long inB, long outB) {
  const float* ip = in + (long)blockIdx.z * inB;
  bf16* op = out + (long)blockIdx.z * outB;
  const int idx = blockIdx.x * blockDim.x + threadIdx.x;
  const int r = idx / C, c = idx % C;
  op[(long)c * R + r] = (bf16)ip[(long)r * C + c];
}

__global__ void layernorm_kernel(float* __restrict__ h, const float* __restrict__ g,
                                 const float* __restrict__ bta, bf16* __restrict__ hb) {
  __shared__ float red[256];
  const int row = blockIdx.x, tid = threadIdx.x;
  float* r = h + (long)row * DMODEL;
  float s = 0.0f;
  for (int i = tid; i < DMODEL; i += 256) s += r[i];
  red[tid] = s; __syncthreads();
  for (int o = 128; o > 0; o >>= 1) { if (tid < o) red[tid] += red[tid + o]; __syncthreads(); }
  const float mu = red[0] / (float)DMODEL;
  __syncthreads();
  float vs = 0.0f;
  for (int i = tid; i < DMODEL; i += 256) { float d = r[i] - mu; vs += d * d; }
  red[tid] = vs; __syncthreads();
  for (int o = 128; o > 0; o >>= 1) { if (tid < o) red[tid] += red[tid + o]; __syncthreads(); }
  const float inv = rsqrtf(red[0] / (float)DMODEL + 1e-5f);
  for (int i = tid; i < DMODEL; i += 256) {
    float v = (r[i] - mu) * inv * g[i] + bta[i];
    r[i] = v;
    hb[(long)row * DMODEL + i] = (bf16)v;
  }
}

// ---------------- host launch ----------------
extern "C" void kernel_launch(void* const* d_in, const int* in_sizes, int n_in,
                              void* d_out, int out_size, void* d_ws, size_t ws_size,
                              hipStream_t stream) {
  const int*   x     = (const int*)d_in[0];
  const float* embed = (const float*)d_in[1];
  const float* pos   = (const float*)d_in[2];
  const float* Qw    = (const float*)d_in[3];
  const float* Kw    = (const float*)d_in[4];
  const float* Vw    = (const float*)d_in[5];
  const float* Ow    = (const float*)d_in[6];
  const float* Win   = (const float*)d_in[7];
  const float* Wout  = (const float*)d_in[8];
  const float* gamma = (const float*)d_in[9];
  const float* beta  = (const float*)d_in[10];
  float* logits = (float*)d_out;

  size_t off = 0;
  auto alloc = [&](size_t bytes) -> char* {
    char* p = (char*)d_ws + off;
    off += (bytes + 255) & ~(size_t)255;
    return p;
  };
  float* h_f32  = (float*)alloc((size_t)BS * DMODEL * 4);
  bf16*  h_bf   = (bf16*) alloc((size_t)BS * DMODEL * 2);
  bf16*  q_bf   = (bf16*) alloc((size_t)BS * DMODEL * 2);
  bf16*  k_bf   = (bf16*) alloc((size_t)BS * DMODEL * 2);
  bf16*  z_bf   = (bf16*) alloc((size_t)BS * DMODEL * 2);
  bf16*  xvT_bf = (bf16*) alloc((size_t)NHEAD * DHEAD * BS * 2);
  bf16*  a1_bf  = (bf16*) alloc((size_t)BS * DMLP * 2);
  bf16*  emb_bf = (bf16*) alloc((size_t)VOCAB * DMODEL * 2);
  bf16*  wq_t   = (bf16*) alloc((size_t)DMODEL * DMODEL * 2);
  bf16*  wk_t   = (bf16*) alloc((size_t)DMODEL * DMODEL * 2);
  bf16*  wv_t   = (bf16*) alloc((size_t)NHEAD * DHEAD * DMODEL * 2);
  bf16*  wo_t   = (bf16*) alloc((size_t)DMODEL * DMODEL * 2);
  bf16*  win_t  = (bf16*) alloc((size_t)DMLP * DMODEL * 2);
  bf16*  wout_t = (bf16*) alloc((size_t)DMODEL * DMLP * 2);
  (void)ws_size; (void)n_in; (void)in_sizes; (void)out_size;

  // embed table to bf16 (used as BT for logits: logits = h @ embed^T)
  cvt_kernel<<<dim3(16384), 256, 0, stream>>>(embed, emb_bf, (long)VOCAB * DMODEL);
  // h = embed[x] + pos
  embed_kernel<<<dim3(BS), 256, 0, stream>>>(x, embed, pos, h_f32, h_bf);

  for (int i = 0; i < NLAYERS; ++i) {
    const float* Qi = Qw   + (size_t)i * DMODEL * DMODEL;
    const float* Ki = Kw   + (size_t)i * DMODEL * DMODEL;
    const float* Vi = Vw   + (size_t)i * NHEAD * DMODEL * DHEAD;
    const float* Oi = Ow   + (size_t)i * DMODEL * DMODEL;
    const float* Wi = Win  + (size_t)i * DMODEL * DMLP;
    const float* Wo = Wout + (size_t)i * DMLP * DMODEL;

    // transpose+convert weights -> BT layout
    tcvt_kernel<<<dim3(DMODEL * DMODEL / 256), 256, 0, stream>>>(Qi, wq_t, DMODEL, DMODEL, 0, 0);
    tcvt_kernel<<<dim3(DMODEL * DMODEL / 256), 256, 0, stream>>>(Ki, wk_t, DMODEL, DMODEL, 0, 0);
    tcvt_kernel<<<dim3(DMODEL * DHEAD / 256, 1, NHEAD), 256, 0, stream>>>(
        Vi, wv_t, DMODEL, DHEAD, (long)DMODEL * DHEAD, (long)DHEAD * DMODEL);
    tcvt_kernel<<<dim3(DMODEL * DMODEL / 256), 256, 0, stream>>>(Oi, wo_t, DMODEL, DMODEL, 0, 0);
    tcvt_kernel<<<dim3(DMODEL * DMLP / 256), 256, 0, stream>>>(Wi, win_t, DMODEL, DMLP, 0, 0);
    tcvt_kernel<<<dim3(DMLP * DMODEL / 256), 256, 0, stream>>>(Wo, wout_t, DMLP, DMODEL, 0, 0);

    // q = h @ Q, k = h @ K  (bf16 outputs)
    gemm_wmma<4, false, false><<<dim3(DMODEL / 64, BS / 64, 1), 32, 0, stream>>>(
        h_bf, DMODEL, 0, wq_t, DMODEL, 0, nullptr, q_bf, DMODEL, 0,
        nullptr, 0, 0, nullptr, DMODEL);
    gemm_wmma<4, false, false><<<dim3(DMODEL / 64, BS / 64, 1), 32, 0, stream>>>(
        h_bf, DMODEL, 0, wk_t, DMODEL, 0, nullptr, k_bf, DMODEL, 0,
        nullptr, 0, 0, nullptr, DMODEL);
    // xvT[h] = (h @ V[h])^T  written directly transposed -> [H][Dh][BS] bf16
    gemm_wmma<4, false, false><<<dim3(1, BS / 64, NHEAD), 32, 0, stream>>>(
        h_bf, DMODEL, 0, wv_t, DMODEL, (long)DHEAD * DMODEL,
        nullptr, nullptr, DHEAD, 0,
        xvT_bf, BS, (long)DHEAD * BS, nullptr, DMODEL);
    // fused causal attention -> z bf16
    flash_attn_kernel<<<dim3(SEQ / 16, NHEAD, BATCH), 32, 0, stream>>>(q_bf, k_bf, xvT_bf, z_bf);
    // h = h + z @ O
    gemm_wmma<4, true, false><<<dim3(DMODEL / 64, BS / 64, 1), 32, 0, stream>>>(
        z_bf, DMODEL, 0, wo_t, DMODEL, 0, h_f32, nullptr, DMODEL, 0,
        nullptr, 0, 0, h_f32, DMODEL);
    // h = LN(h)
    layernorm_kernel<<<dim3(BS), 256, 0, stream>>>(h_f32, gamma + (size_t)i * DMODEL,
                                                   beta + (size_t)i * DMODEL, h_bf);
    // a1 = gelu(h @ W_in)  (bf16)
    gemm_wmma<4, false, true><<<dim3(DMLP / 64, BS / 64, 1), 32, 0, stream>>>(
        h_bf, DMODEL, 0, win_t, DMODEL, 0, nullptr, a1_bf, DMLP, 0,
        nullptr, 0, 0, nullptr, DMODEL);
    // h = h + a1 @ W_out  (emit f32 h and bf16 h together)
    gemm_wmma<4, true, false><<<dim3(DMODEL / 64, BS / 64, 1), 32, 0, stream>>>(
        a1_bf, DMLP, 0, wout_t, DMLP, 0, h_f32, h_bf, DMODEL, 0,
        nullptr, 0, 0, h_f32, DMLP);
  }

  // logits = h @ embed^T  (f32 output)
  gemm_wmma<4, false, false><<<dim3(VOCAB / 64, BS / 64, 1), 32, 0, stream>>>(
      h_bf, DMODEL, 0, emb_bf, DMODEL, 0, logits, nullptr, VOCAB, 0,
      nullptr, 0, 0, nullptr, DMODEL);
}
